// MAGNN_lp_73598559584372
// MI455X (gfx1250) — matrix operand order repaired and checked by reading, not compiled, not run on hardware
//
#include <hip/hip_runtime.h>
#include <hip/hip_bf16.h>

// ---------------------------------------------------------------------------
// MAGNN link prediction forward for gfx1250 (MI455X, wave32, WMMA).
// All dense GEMMs (towers, semantic attention, product MLP) use
// v_wmma_f32_16x16x32_bf16 (bf16 in, fp32 accumulate), LDS-staged.
// ---------------------------------------------------------------------------

#define N0c   20000
#define N1c   20000
#define F0c   512
#define HIDc  64
#define Ec    200000
#define Bc    8192
#define AVc   128
#define CHc   128

typedef __attribute__((ext_vector_type(16))) __bf16 v16bf;
typedef __attribute__((ext_vector_type(8)))  float  v8f;

__device__ __forceinline__ void atomicMaxFloat(float* addr, float v) {
  // Standard monotone-bits trick: works for mixed signs and -inf init.
  if (v >= 0.0f) atomicMax((int*)addr, __float_as_int(v));
  else           atomicMin((unsigned int*)addr, __float_as_uint(v));
}

__device__ __forceinline__ float eluf(float v) {
  return v > 0.0f ? v : expm1f(v);
}

// Load a 16-bit A fragment (ISA 16x32 layout) from an LDS row pointer.
__device__ __forceinline__ v16bf frag_a(const __bf16* arow, int half) {
  v16bf a;
#pragma unroll
  for (int i = 0; i < 8; ++i) a[i] = arow[half * 8 + i];
#pragma unroll
  for (int i = 0; i < 8; ++i) a[8 + i] = arow[16 + half * 8 + i];
  return a;
}
// Load a 16-bit B fragment (ISA 32x16 layout) from a transposed LDS row.
__device__ __forceinline__ v16bf frag_b(const __bf16* brow, int half) {
  v16bf b;
#pragma unroll
  for (int i = 0; i < 16; ++i) b[i] = brow[(i < 8) ? (half * 8 + i) : (16 + half * 8 + (i - 8))];
  return b;
}

// ---------------------------------------------------------------------------
// Kernel 1: fused feature tower.
//   Z = X@PW + pb ; H = gelu(Z) ; Y = H@W2 + b2 + Z ; LayerNorm(Y)*g + be
// One block = 32 node rows. 8 waves, each owns a 16x16 tile of the 64 outs.
// ---------------------------------------------------------------------------
__global__ __launch_bounds__(256) void magnn_tower_kernel(
    const float* __restrict__ feats0, const float* __restrict__ feats1,
    const float* __restrict__ pw0, const float* __restrict__ pb0,
    const float* __restrict__ w20, const float* __restrict__ b20,
    const float* __restrict__ g0,  const float* __restrict__ be0,
    const float* __restrict__ pw1, const float* __restrict__ pb1,
    const float* __restrict__ w21, const float* __restrict__ b21,
    const float* __restrict__ g1,  const float* __restrict__ be1,
    float* __restrict__ feat)
{
  __shared__ __align__(16) __bf16 sA [32][40];  // A chunk 32 rows x 32 k (bf16)
  __shared__ __align__(16) __bf16 sBt[64][40];  // B chunk transposed: [n][k0..31]
  __shared__ __align__(16) __bf16 hS [32][72];  // gelu(Z) for GEMM2, [m][k0..63]
  __shared__ __align__(16) __bf16 w2T[64][72];  // W2 transposed: [n][k0..63]
  __shared__ float zS[32][65];                  // Z, later pre-LN Y

  const int t    = threadIdx.x;
  const int lane = t & 31;
  const int wave = t >> 5;
  const int mt   = wave >> 2;      // row tile 0..1
  const int nt   = wave & 3;       // col tile 0..3
  const int nl   = lane & 15;
  const int half = lane >> 4;

  const int blockRow = blockIdx.x * 32;
  const int tower = (blockRow >= N0c) ? 1 : 0;
  const float* X  = tower ? feats1 : feats0;
  const int xrow0 = blockRow - tower * N0c;
  const float* pw = tower ? pw1 : pw0;
  const float* pb = tower ? pb1 : pb0;
  const float* w2 = tower ? w21 : w20;
  const float* b2 = tower ? b21 : b20;
  const float* g  = tower ? g1  : g0;
  const float* be = tower ? be1 : be0;

  // Preload W2 (64x64) transposed as bf16 (used after the main K loop).
  {
    const int r = t >> 2, cb = (t & 3) * 16;
#pragma unroll
    for (int i = 0; i < 16; ++i)
      w2T[cb + i][r] = (__bf16)w2[r * 64 + cb + i];
  }

  v8f c = {};
  for (int kc = 0; kc < 16; ++kc) {             // K = 512 in chunks of 32
    {  // stage A chunk: 32 rows x 32 k, one float4 per thread
      const int r = t >> 3, c4 = (t & 7) * 4;
      const float4 f = *(const float4*)(X + (size_t)(xrow0 + r) * F0c + kc * 32 + c4);
      sA[r][c4 + 0] = (__bf16)f.x; sA[r][c4 + 1] = (__bf16)f.y;
      sA[r][c4 + 2] = (__bf16)f.z; sA[r][c4 + 3] = (__bf16)f.w;
      if (kc < 15)  // CDNA5 prefetch of next A panel (global_prefetch_b8)
        __builtin_prefetch(X + (size_t)(xrow0 + r) * F0c + (kc + 1) * 32 + c4, 0, 0);
    }
    {  // stage B chunk transposed: 32 k x 64 n, two float4 per thread
      const int r = t >> 3, c8 = (t & 7) * 8;
      const float4 fa = *(const float4*)(pw + (size_t)(kc * 32 + r) * 64 + c8);
      const float4 fb = *(const float4*)(pw + (size_t)(kc * 32 + r) * 64 + c8 + 4);
      sBt[c8 + 0][r] = (__bf16)fa.x; sBt[c8 + 1][r] = (__bf16)fa.y;
      sBt[c8 + 2][r] = (__bf16)fa.z; sBt[c8 + 3][r] = (__bf16)fa.w;
      sBt[c8 + 4][r] = (__bf16)fb.x; sBt[c8 + 5][r] = (__bf16)fb.y;
      sBt[c8 + 6][r] = (__bf16)fb.z; sBt[c8 + 7][r] = (__bf16)fb.w;
    }
    __syncthreads();
    const v16bf a = frag_a(&sA[mt * 16 + nl][0], half);
    const v16bf b = frag_b(&sBt[nt * 16 + nl][0], half);
    c = __builtin_amdgcn_wmma_f32_16x16x32_bf16(false, a, false, b, (short)0, c,
                                                false, false);
    __syncthreads();
  }

  // Epilogue 1: bias + exact GELU; stash Z (fp32) and gelu(Z) (bf16).
  const int col = nt * 16 + nl;
  {
    const float bias = pb[col];
#pragma unroll
    for (int v = 0; v < 8; ++v) {
      const int row = mt * 16 + half * 8 + v;
      const float z = c[v] + bias;
      const float h = 0.5f * z * (1.0f + erff(z * 0.70710678118654752f));
      zS[row][col] = z;
      hS[row][col] = (__bf16)h;
    }
  }
  __syncthreads();

  // GEMM2: Y2 = gelu(Z)[32x64] @ W2[64x64], K = 64 in two chunks of 32.
  v8f c2 = {};
#pragma unroll
  for (int kc = 0; kc < 2; ++kc) {
    const v16bf a = frag_a(&hS[mt * 16 + nl][kc * 32], half);
    const v16bf b = frag_b(&w2T[nt * 16 + nl][kc * 32], half);
    c2 = __builtin_amdgcn_wmma_f32_16x16x32_bf16(false, a, false, b, (short)0, c2,
                                                 false, false);
  }

  // Epilogue 2: residual, then LayerNorm per row (64 elements).
  {
    const float bias2 = b2[col];
#pragma unroll
    for (int v = 0; v < 8; ++v) {
      const int row = mt * 16 + half * 8 + v;
      zS[row][col] = c2[v] + bias2 + zS[row][col];
    }
  }
  __syncthreads();
  {
    const int r = t >> 3, s8 = (t & 7) * 8;   // 8 threads per row
    float sum = 0.0f, sq = 0.0f;
#pragma unroll
    for (int i = 0; i < 8; ++i) { const float y = zS[r][s8 + i]; sum += y; sq += y * y; }
    sum += __shfl_xor(sum, 1); sq += __shfl_xor(sq, 1);
    sum += __shfl_xor(sum, 2); sq += __shfl_xor(sq, 2);
    sum += __shfl_xor(sum, 4); sq += __shfl_xor(sq, 4);
    const float mu   = sum * (1.0f / 64.0f);
    const float var  = sq * (1.0f / 64.0f) - mu * mu;
    const float rstd = rsqrtf(var + 1e-5f);
    const size_t orow = (size_t)(blockRow + r) * HIDc;
#pragma unroll
    for (int i = 0; i < 8; ++i) {
      const int cc = s8 + i;
      feat[orow + cc] = (zS[r][cc] - mu) * rstd * g[cc] + be[cc];
    }
  }
}

// ---------------------------------------------------------------------------
// Kernel 2: cumulative complex rotations for the two etype configs.
// cfg 0 = user [0,1], cfg 1 = item [1,0]; rfull[0]=norm(r0), rfull[1]=conj.
// ---------------------------------------------------------------------------
__global__ void magnn_rot_kernel(const float* __restrict__ r_vec,
                                 float* __restrict__ frbuf)
{
  const int t = threadIdx.x;
  if (t >= 64) return;
  const int cfg = t >> 5, j = t & 31;
  float rr = r_vec[j * 2 + 0], ri = r_vec[j * 2 + 1];   // r_vec[0][j]
  const float inv = rsqrtf(rr * rr + ri * ri);
  rr *= inv; ri *= inv;
  const int et0 = (cfg == 0) ? 0 : 1;
  const int et1 = (cfg == 0) ? 1 : 0;
  // fr[2] = identity
  const float rv1r = rr, rv1i = (et1 == 0) ? ri : -ri;
  const float f1r = rv1r, f1i = rv1i;                   // fr[1] = fr[2]*rv1
  const float rv0r = rr, rv0i = (et0 == 0) ? ri : -ri;
  const float f0r = f1r * rv0r - f1i * rv0i;            // fr[0] = fr[1]*rv0
  const float f0i = f1r * rv0i + f1i * rv0r;
  float* fb = frbuf + cfg * (3 * 32 * 2);
  fb[(0 * 32 + j) * 2 + 0] = f0r; fb[(0 * 32 + j) * 2 + 1] = f0i;
  fb[(1 * 32 + j) * 2 + 0] = f1r; fb[(1 * 32 + j) * 2 + 1] = f1i;
  fb[(2 * 32 + j) * 2 + 0] = 1.0f; fb[(2 * 32 + j) * 2 + 1] = 0.0f;
}

__global__ void magnn_fill_kernel(float* __restrict__ p, float v, size_t n) {
  const size_t i = (size_t)blockIdx.x * blockDim.x + threadIdx.x;
  if (i < n) p[i] = v;
}

// ---------------------------------------------------------------------------
// Kernel 3a: gather + rotate + mean + attention score + segment max.
// One wave per metapath instance; lane j owns complex pair j (dims 2j,2j+1).
// ---------------------------------------------------------------------------
__global__ __launch_bounds__(256) void magnn_meta_hidden_kernel(
    const float* __restrict__ feat, const int* __restrict__ emi,
    const int* __restrict__ tgt, const float* __restrict__ attn,
    const float* __restrict__ fr, float* __restrict__ hidden,
    float* __restrict__ escore, float* __restrict__ segmax)
{
  const int e = blockIdx.x * 8 + (threadIdx.x >> 5);
  const int j = threadIdx.x & 31;
  if (e >= Ec) return;
  float ar = 0.0f, ai = 0.0f;
#pragma unroll
  for (int l = 0; l < 3; ++l) {
    const int node = emi[(size_t)e * 3 + l];
    const float* fp = feat + (size_t)node * HIDc + j * 2;
    const float re = fp[0], im = fp[1];
    const float frr = fr[(l * 32 + j) * 2 + 0];
    const float fri = fr[(l * 32 + j) * 2 + 1];
    ar += re * frr - im * fri;
    ai += re * fri + im * frr;
  }
  ar *= (1.0f / 3.0f); ai *= (1.0f / 3.0f);
  hidden[(size_t)e * HIDc + j * 2 + 0] = ar;
  hidden[(size_t)e * HIDc + j * 2 + 1] = ai;
  const int h = j >> 2, d0 = (j & 3) * 2;                 // head, dim-in-head
  float part = ar * attn[h * 8 + d0] + ai * attn[h * 8 + d0 + 1];
  part += __shfl_xor(part, 1);
  part += __shfl_xor(part, 2);
  if ((j & 3) == 0) {
    const float s = part > 0.0f ? part : 0.01f * part;    // leaky_relu(0.01)
    escore[(size_t)e * 8 + h] = s;
    atomicMaxFloat(&segmax[tgt[e] * 8 + h], s);
  }
}

// Kernel 3b: a = exp(e - max), accumulate segment sums.
__global__ void magnn_softmax_kernel(const int* __restrict__ tgt,
                                     float* __restrict__ escore,
                                     const float* __restrict__ segmax,
                                     float* __restrict__ ssum)
{
  const size_t i = (size_t)blockIdx.x * blockDim.x + threadIdx.x;
  if (i >= (size_t)Ec * 8) return;
  const int e = (int)(i >> 3), h = (int)(i & 7);
  const float a = expf(escore[i] - segmax[tgt[e] * 8 + h]);
  escore[i] = a;
  atomicAdd(&ssum[tgt[e] * 8 + h], a);
}

// Kernel 3c: weighted scatter-add of instance features into per-target slots.
__global__ void magnn_aggregate_kernel(const int* __restrict__ tgt,
                                       const float* __restrict__ escore,
                                       const float* __restrict__ ssum,
                                       const float* __restrict__ hidden,
                                       float* __restrict__ ret)
{
  const size_t i = (size_t)blockIdx.x * blockDim.x + threadIdx.x;
  if (i >= (size_t)Ec * 64) return;
  const int e = (int)(i >> 6), d = (int)(i & 63), h = d >> 3;
  const int tg = tgt[e];
  const float w = escore[(size_t)e * 8 + h] / (ssum[tg * 8 + h] + 1e-9f);
  atomicAdd(&ret[(size_t)tg * 64 + d], w * hidden[i]);
}

// ---------------------------------------------------------------------------
// Kernel 4 (WMMA): semantic attention score
//   sacc += sum_b tanh(elu(ret_b) @ w1 + b1) @ w2    (mean taken later)
// One block = 16 rows of B; 8 waves each own a 16x16 tile of the 128 AV cols.
// ---------------------------------------------------------------------------
__global__ __launch_bounds__(256) void magnn_semantic_score_kernel(
    const float* __restrict__ ret, const float* __restrict__ w1,
    const float* __restrict__ b1, const float* __restrict__ w2,
    float* __restrict__ sacc)
{
  __shared__ __align__(16) __bf16 aS [16][72];    // elu rows, [m][k0..63]
  __shared__ __align__(16) __bf16 w1T[128][72];   // w1 transposed: [n][k0..63]
  __shared__ float red[8];

  const int t = threadIdx.x;
  const int lane = t & 31, wave = t >> 5;
  const int nl = lane & 15, half = lane >> 4;
  const int rowBase = blockIdx.x * 16;

  {  // stage elu(ret) rows: 16x64, 4 per thread
    const int r = t >> 4, c4 = (t & 15) * 4;
#pragma unroll
    for (int i = 0; i < 4; ++i)
      aS[r][c4 + i] = (__bf16)eluf(ret[(size_t)(rowBase + r) * HIDc + c4 + i]);
  }
  {  // stage w1 transposed: 64x128, 32 per thread
    const int kr = t >> 2, cb = (t & 3) * 32;
#pragma unroll
    for (int i = 0; i < 32; ++i)
      w1T[cb + i][kr] = (__bf16)w1[kr * AVc + cb + i];
  }
  __syncthreads();

  v8f c = {};
#pragma unroll
  for (int kc = 0; kc < 2; ++kc) {
    const v16bf a = frag_a(&aS[nl][kc * 32], half);
    const v16bf b = frag_b(&w1T[wave * 16 + nl][kc * 32], half);
    c = __builtin_amdgcn_wmma_f32_16x16x32_bf16(false, a, false, b, (short)0, c,
                                                false, false);
  }

  // Epilogue: tanh(. + b1[col]) * w2[col], full sum over the 16x128 tile.
  const int col = wave * 16 + nl;
  const float bb = b1[col], ww = w2[col];
  float part = 0.0f;
#pragma unroll
  for (int v = 0; v < 8; ++v) part += tanhf(c[v] + bb) * ww;
  part += __shfl_xor(part, 1);
  part += __shfl_xor(part, 2);
  part += __shfl_xor(part, 4);
  part += __shfl_xor(part, 8);
  part += __shfl_xor(part, 16);
  if (lane == 0) red[wave] = part;
  __syncthreads();
  if (t == 0) {
    float tot = 0.0f;
#pragma unroll
    for (int i = 0; i < 8; ++i) tot += red[i];
    atomicAdd(sacc, tot);
  }
}

__global__ void magnn_beta_kernel(const float* __restrict__ sacc,
                                  float* __restrict__ beta)
{
  if (threadIdx.x == 0) {
    for (int s = 0; s < 2; ++s) {
      const float a = sacc[s * 2 + 0] * (1.0f / Bc);
      const float b = sacc[s * 2 + 1] * (1.0f / Bc);
      const float m = fmaxf(a, b);
      const float ea = expf(a - m), eb = expf(b - m);
      const float inv = 1.0f / (ea + eb);
      beta[s * 2 + 0] = ea * inv;
      beta[s * 2 + 1] = eb * inv;
    }
  }
}

// Kernel 5: h_user*h_item (semantic-weighted elu combos).
__global__ void magnn_combine_kernel(
    const float* __restrict__ ru0, const float* __restrict__ ru1,
    const float* __restrict__ ri0, const float* __restrict__ ri1,
    const float* __restrict__ beta, float* __restrict__ xbuf)
{
  const size_t i = (size_t)blockIdx.x * blockDim.x + threadIdx.x;
  if (i >= (size_t)Bc * 64) return;
  const float hu = beta[0] * eluf(ru0[i]) + beta[1] * eluf(ru1[i]);
  const float hi = beta[2] * eluf(ri0[i]) + beta[3] * eluf(ri1[i]);
  xbuf[i] = hu * hi;
}

// ---------------------------------------------------------------------------
// Kernel 6 (WMMA): product MLP + row softmax.
//   h = relu(x @ cw1 + cb1) ; logits = h @ cw2 ; softmax rows
// One block = 16 rows; 8 waves tile the 128 hidden cols.
// ---------------------------------------------------------------------------
__global__ __launch_bounds__(256) void magnn_mlp_kernel(
    const float* __restrict__ xbuf, const float* __restrict__ cw1,
    const float* __restrict__ cb1, const float* __restrict__ cw2,
    float* __restrict__ out)
{
  __shared__ __align__(16) __bf16 xS  [16][72];   // x rows, [m][k0..63]
  __shared__ __align__(16) __bf16 cw1T[128][72];  // cw1 transposed: [n][k0..63]
  __shared__ float hbuf[16][132];                 // relu hidden, fp32

  const int t = threadIdx.x;
  const int lane = t & 31, wave = t >> 5;
  const int nl = lane & 15, half = lane >> 4;
  const int rowBase = blockIdx.x * 16;

  {  // stage x rows: 16x64, 4 per thread
    const int r = t >> 4, c4 = (t & 15) * 4;
#pragma unroll
    for (int i = 0; i < 4; ++i)
      xS[r][c4 + i] = (__bf16)xbuf[(size_t)(rowBase + r) * HIDc + c4 + i];
  }
  {  // stage cw1 transposed: 64x128, 32 per thread
    const int kr = t >> 2, cb = (t & 3) * 32;
#pragma unroll
    for (int i = 0; i < 32; ++i)
      cw1T[cb + i][kr] = (__bf16)cw1[kr * CHc + cb + i];
  }
  __syncthreads();

  v8f c = {};
#pragma unroll
  for (int kc = 0; kc < 2; ++kc) {
    const v16bf a = frag_a(&xS[nl][kc * 32], half);
    const v16bf b = frag_b(&cw1T[wave * 16 + nl][kc * 32], half);
    c = __builtin_amdgcn_wmma_f32_16x16x32_bf16(false, a, false, b, (short)0, c,
                                                false, false);
  }

  const int col = wave * 16 + nl;
  const float bb = cb1[col];
#pragma unroll
  for (int v = 0; v < 8; ++v)
    hbuf[half * 8 + v][col] = fmaxf(c[v] + bb, 0.0f);
  __syncthreads();

  if (t < 16) {  // per-row 128-dim dot with cw2, then 2-way softmax
    float l0 = 0.0f, l1 = 0.0f;
    for (int k = 0; k < CHc; ++k) {
      const float hv = hbuf[t][k];
      l0 += hv * cw2[k * 2 + 0];
      l1 += hv * cw2[k * 2 + 1];
    }
    const float m = fmaxf(l0, l1);
    const float e0 = expf(l0 - m), e1 = expf(l1 - m);
    const float inv = 1.0f / (e0 + e1);
    out[(rowBase + t) * 2 + 0] = e0 * inv;
    out[(rowBase + t) * 2 + 1] = e1 * inv;
  }
}

// ---------------------------------------------------------------------------
extern "C" void kernel_launch(void* const* d_in, const int* in_sizes, int n_in,
                              void* d_out, int out_size, void* d_ws, size_t ws_size,
                              hipStream_t stream)
{
  (void)in_sizes; (void)n_in; (void)out_size; (void)ws_size;
  const float* feats0 = (const float*)d_in[0];
  const float* feats1 = (const float*)d_in[1];
  const float* t0pw = (const float*)d_in[2];
  const float* t0pb = (const float*)d_in[3];
  const float* t0w2 = (const float*)d_in[4];
  const float* t0b2 = (const float*)d_in[5];
  const float* t0g  = (const float*)d_in[6];
  const float* t0be = (const float*)d_in[7];
  const float* t1pw = (const float*)d_in[8];
  const float* t1pb = (const float*)d_in[9];
  const float* t1w2 = (const float*)d_in[10];
  const float* t1b2 = (const float*)d_in[11];
  const float* t1g  = (const float*)d_in[12];
  const float* t1be = (const float*)d_in[13];
  const float* r_vec     = (const float*)d_in[14];
  const float* attn_user = (const float*)d_in[15];
  const float* attn_item = (const float*)d_in[16];
  const float* su_w1 = (const float*)d_in[17];
  const float* su_b1 = (const float*)d_in[18];
  const float* su_w2 = (const float*)d_in[19];
  const float* si_w1 = (const float*)d_in[20];
  const float* si_b1 = (const float*)d_in[21];
  const float* si_w2 = (const float*)d_in[22];
  const float* cw1 = (const float*)d_in[23];
  const float* cb1 = (const float*)d_in[24];
  const float* cw2 = (const float*)d_in[25];
  // d_in[26]=idx0, d_in[27]=idx1: identity concat (0..N0-1, N0..N0+N1-1).
  const int* emi_user = (const int*)d_in[28];
  const int* tgt_user = (const int*)d_in[29];
  const int* emi_item = (const int*)d_in[30];
  const int* tgt_item = (const int*)d_in[31];
  float* out = (float*)d_out;

  // Workspace carve-out (floats, 64-element aligned).
  float* w = (float*)d_ws;
  size_t off = 0;
  auto alloc = [&](size_t n) { float* p = w + off; off += (n + 63) & ~(size_t)63; return p; };
  float* feat   = alloc((size_t)(N0c + N1c) * HIDc);  // node embedding table
  float* frbuf  = alloc(2 * 3 * 32 * 2);              // rotations per cfg
  float* segmax = alloc((size_t)Bc * 8);
  float* ssum   = alloc((size_t)Bc * 8);
  float* escore = alloc((size_t)Ec * 8);
  float* hidden = alloc((size_t)Ec * HIDc);
  float* ret    = alloc((size_t)4 * Bc * HIDc);       // [user0,user1,item0,item1]
  float* sacc   = alloc(4);
  float* beta   = alloc(4);
  float* xbuf   = alloc((size_t)Bc * HIDc);

  // Phase 1: feature towers -> feat[(N0+N1) x 64]  (WMMA bf16).
  magnn_tower_kernel<<<(N0c + N1c) / 32, 256, 0, stream>>>(
      feats0, feats1, t0pw, t0pb, t0w2, t0b2, t0g, t0be,
      t1pw, t1pb, t1w2, t1b2, t1g, t1be, feat);

  // Phase 2: rotations + zero accumulators.
  magnn_rot_kernel<<<1, 64, 0, stream>>>(r_vec, frbuf);
  magnn_fill_kernel<<<(int)(((size_t)4 * Bc * HIDc + 255) / 256), 256, 0, stream>>>(
      ret, 0.0f, (size_t)4 * Bc * HIDc);
  magnn_fill_kernel<<<1, 64, 0, stream>>>(sacc, 0.0f, 4);

  // Phase 3: four metapaths (user p0,p1 then item p0,p1), buffers reused.
  for (int mp = 0; mp < 4; ++mp) {
    const int side = mp >> 1, p = mp & 1;
    const int* emi = (side == 0 ? emi_user : emi_item) + (size_t)p * Ec * 3;
    const int* tgt = (side == 0 ? tgt_user : tgt_item) + (size_t)p * Ec;
    const float* attn = (side == 0 ? attn_user : attn_item) + p * 64;
    const float* fr = frbuf + side * (3 * 32 * 2);
    float* retP = ret + (size_t)mp * Bc * HIDc;

    magnn_fill_kernel<<<(Bc * 8 + 255) / 256, 256, 0, stream>>>(
        segmax, -__builtin_inff(), (size_t)Bc * 8);
    magnn_fill_kernel<<<(Bc * 8 + 255) / 256, 256, 0, stream>>>(
        ssum, 0.0f, (size_t)Bc * 8);
    magnn_meta_hidden_kernel<<<Ec / 8, 256, 0, stream>>>(
        feat, emi, tgt, attn, fr, hidden, escore, segmax);
    magnn_softmax_kernel<<<(int)(((size_t)Ec * 8 + 255) / 256), 256, 0, stream>>>(
        tgt, escore, segmax, ssum);
    magnn_aggregate_kernel<<<(int)(((size_t)Ec * 64 + 255) / 256), 256, 0, stream>>>(
        tgt, escore, ssum, hidden, retP);
  }

  // Phase 4: semantic attention (WMMA) -> beta.
  for (int mp = 0; mp < 4; ++mp) {
    const int side = mp >> 1;
    const float* w1 = side ? si_w1 : su_w1;
    const float* b1 = side ? si_b1 : su_b1;
    const float* w2 = side ? si_w2 : su_w2;
    magnn_semantic_score_kernel<<<Bc / 16, 256, 0, stream>>>(
        ret + (size_t)mp * Bc * HIDc, w1, b1, w2, sacc + mp);
  }
  magnn_beta_kernel<<<1, 32, 0, stream>>>(sacc, beta);
  magnn_combine_kernel<<<(Bc * HIDc + 255) / 256, 256, 0, stream>>>(
      ret, ret + (size_t)Bc * HIDc, ret + (size_t)2 * Bc * HIDc,
      ret + (size_t)3 * Bc * HIDc, beta, xbuf);

  // Phase 5: product MLP + softmax (WMMA) -> out[B x 2].
  magnn_mlp_kernel<<<Bc / 16, 256, 0, stream>>>(xbuf, cw1, cb1, cw2, out);
}
